// TextRNN_63797444215446
// MI455X (gfx1250) — compile-verified, matrix operand order checked by reading, not run on hardware
//
#include <hip/hip_runtime.h>

typedef _Float16 v16h __attribute__((ext_vector_type(16)));
typedef _Float16 v8h  __attribute__((ext_vector_type(8)));
typedef float    v8f  __attribute__((ext_vector_type(8)));

#define B_ 64
#define S_ 512
#define E_ 768
#define H_ 256
#define G4 1024  /* 4*H */

// Fragment K-permutation within a 32-wide K block (ISA §7.12.2):
// lane-half 0 wants K {0..7,16..23}; lane-half 1 wants K {8..15,24..31}.
// Stored order per block: groups {0,2,1,3} -> contiguous v16h per lane-half.
__device__ __forceinline__ int permk(int k) {  // k in [0,32)
  int kg = (k >> 3) & 3;
  int pg = (kg == 1) ? 2 : ((kg == 2) ? 1 : kg);
  return pg * 8 + (k & 7);
}

// ---------------------------------------------------------------------------
// CDNA5 async memory->LDS copy (16 B per lane), tracked by ASYNCcnt.
// LDS address = low 32 bits of the generic shared-space pointer (aperture
// rule: for LDS space, ADDR[31:0] is the LDS byte address).
// ---------------------------------------------------------------------------
__device__ __forceinline__ void async_b128(void* lds_dst, const void* gsrc) {
  asm volatile("global_load_async_to_lds_b128 %0, %1, off"
               :: "v"((unsigned)(uintptr_t)lds_dst), "v"(gsrc)
               : "memory");
}
__device__ __forceinline__ void wait_async0() {
  asm volatile("s_wait_asynccnt 0x0" ::: "memory");
}

// ---------------------------------------------------------------------------
// fp32 -> f16 conversion (grid-stride) — for x
// ---------------------------------------------------------------------------
__global__ void k_f32_to_f16(const float* __restrict__ in,
                             _Float16* __restrict__ out, int n) {
  int i = blockIdx.x * blockDim.x + threadIdx.x;
  int stride = gridDim.x * blockDim.x;
  for (; i < n; i += stride) out[i] = (_Float16)in[i];
}

// ---------------------------------------------------------------------------
// One-time weight prep: in[K][1024] f32 -> out[1024][K] f16, with each
// 32-wide K block internally permuted into WMMA B-fragment order.
// ---------------------------------------------------------------------------
__global__ void k_w_to_f16T(const float* __restrict__ in,
                            _Float16* __restrict__ out, int K) {
  int idx = blockIdx.x * blockDim.x + threadIdx.x;
  const int total = K * G4;
  const int stride = gridDim.x * blockDim.x;
  for (; idx < total; idx += stride) {
    int k = idx >> 10;        // row (coalesced read over n)
    int n = idx & 1023;
    int kcol = (k & ~31) + permk(k & 31);
    out[(size_t)n * K + kcol] = (_Float16)in[idx];
  }
}

// ---------------------------------------------------------------------------
// xz = x @ kernel + bias  (M=32768, K=768, N=1024, per direction)
// block = 128 threads (4 waves); block tile 64x64; wave tile 16x64
// A staged permuted; B from pre-transposed kT; staging via async-to-LDS.
// ---------------------------------------------------------------------------
__global__ __launch_bounds__(128) void k_gemm_xz(
    const _Float16* __restrict__ x16,   // [B*S, E]
    const _Float16* __restrict__ kT,    // [2][1024][E] permuted
    const float* __restrict__ bias_f,
    const float* __restrict__ bias_b,
    _Float16* __restrict__ xz_f,        // [B*S, 4H] f16
    _Float16* __restrict__ xz_b) {
  __shared__ __align__(32) _Float16 As[64 * 48];  // LDA=48 (96B, 32B-mult)
  __shared__ __align__(32) _Float16 Bt[64 * 48];

  const int dir = blockIdx.z;
  const _Float16* kw = kT + (size_t)dir * G4 * E_;
  const float* bias = dir ? bias_b : bias_f;
  _Float16* xz = dir ? xz_b : xz_f;

  const int m0 = blockIdx.y * 64;
  const int n0 = blockIdx.x * 64;
  const int tid = threadIdx.x;
  const int lane = tid & 31;
  const int wave = tid >> 5;
  const int l15 = lane & 15;
  const int lhalf = lane >> 4;

  v8f acc[4] = {};

  for (int k0 = 0; k0 < E_; k0 += 32) {
    // stage A (permuted): 2 threads/row, two async 16B chunks each
    {
      const int r = tid >> 1, half = tid & 1;
      const _Float16* src = x16 + (size_t)(m0 + r) * E_ + k0 + half * 16;
      // half0: K{0..7}->lds 0, K{8..15}->lds 16 ; half1: K{16..23}->8, K{24..31}->24
      async_b128(&As[r * 48 + (half ? 8 : 0)], src);
      async_b128(&As[r * 48 + (half ? 24 : 16)], src + 8);
      if (k0 + 32 < E_) __builtin_prefetch(src + 32, 0, 3);
    }
    // stage B from pre-permuted kT: async vector copy (64 rows x 32 halfs)
    for (int i = 0; i < 2; ++i) {
      const int c = tid * 2 + i;         // 0..255
      const int n = c >> 2;
      const int j = (c & 3) * 8;
      const _Float16* src = kw + (size_t)(n0 + n) * E_ + k0 + j;
      async_b128(&Bt[n * 48 + j], src);
      if (k0 + 32 < E_) __builtin_prefetch(src + 32, 0, 3);
    }
    wait_async0();
    __syncthreads();

    const v16h a = *(const v16h*)&As[(wave * 16 + l15) * 48 + lhalf * 16];
    for (int nt = 0; nt < 4; ++nt) {
      const v16h b = *(const v16h*)&Bt[(nt * 16 + l15) * 48 + lhalf * 16];
      acc[nt] = __builtin_amdgcn_wmma_f32_16x16x32_f16(
          false, a, false, b, (short)0, acc[nt], false, false);
    }
    __syncthreads();
  }

  // store (f16) with bias; C layout: row = r + 8*lhalf, col = l15
  const int mbase = m0 + wave * 16 + lhalf * 8;
  for (int nt = 0; nt < 4; ++nt) {
    const int col = n0 + nt * 16 + l15;
    const float bv = bias[col];
    for (int r = 0; r < 8; ++r)
      xz[(size_t)(mbase + r) * G4 + col] = (_Float16)(acc[nt][r] + bv);
  }
}

// ---------------------------------------------------------------------------
// Persistent recurrent kernel. Grid (4 batch-tiles, 2 dirs), 256 thr (8 waves).
// z[16,1024] = xz[t] + h16 @ rec ; wave w owns h-columns [w*32,w*32+32) of
// ALL FOUR gates -> gate combine stays in registers. h16 kept in the
// permuted A-fragment layout so the next step's A is one v16h LDS load.
// rec K-slices staged into LDS with async-to-LDS (ASYNCcnt).
// ---------------------------------------------------------------------------
__global__ __launch_bounds__(256) void k_lstm(
    const _Float16* __restrict__ rT,     // [2][1024][H] permuted
    const _Float16* __restrict__ xz_f,   // [B*S, 4H] f16
    const _Float16* __restrict__ xz_b,
    float* __restrict__ h_f,             // [B, S, H]
    float* __restrict__ h_b) {
  extern __shared__ __align__(32) char smem[];
  _Float16* h16  = (_Float16*)smem;                   // [16][256] 8 KB (permuted)
  _Float16* recT = (_Float16*)(smem + 16 * 256 * 2);  // [1024][48] 96 KB
  const int LDR = 48;

  const int dir = blockIdx.y;
  const int b0 = blockIdx.x * 16;
  const _Float16* rec = rT + (size_t)dir * G4 * H_;
  const _Float16* xz = dir ? xz_b : xz_f;
  float* hout = dir ? h_b : h_f;

  const int tid = threadIdx.x;
  const int lane = tid & 31;
  const int wave = tid >> 5;
  const int lhalf = lane >> 4;
  const int l15 = lane & 15;
  const int jcol0 = wave * 32;  // this wave's h-column slice

  for (int i = tid; i < 16 * 256; i += 256) h16[i] = (_Float16)0.f;
  __syncthreads();

  float creg[16];
  for (int i = 0; i < 16; ++i) creg[i] = 0.f;

  for (int ss = 0; ss < S_; ++ss) {
    const int t = dir ? (S_ - 1 - ss) : ss;

    // acc[g][jt] initialized from xz (z = xz + h@rec); bias already folded in
    v8f acc[4][2];
    for (int g = 0; g < 4; ++g)
      for (int jt = 0; jt < 2; ++jt) {
        const int col = g * H_ + jcol0 + jt * 16 + l15;
        const size_t base = ((size_t)(b0 + lhalf * 8) * S_ + t) * G4 + col;
        v8f v;
        for (int r = 0; r < 8; ++r)
          v[r] = (float)xz[base + (size_t)r * S_ * G4];
        acc[g][jt] = v;
      }

    for (int k0 = 0; k0 < H_; k0 += 32) {
      // stage rec K-slice [k0,k0+32) x 1024 n (pre-permuted): async copy
      for (int it = 0; it < 16; ++it) {
        const int c = it * 256 + tid;     // 0..4095
        const int n = c >> 2;
        const int j = (c & 3) * 8;
        async_b128(&recT[n * LDR + j], rec + (size_t)n * H_ + k0 + j);
      }
      wait_async0();
      __syncthreads();

      // A fragment from permuted h16: single contiguous v16h
      const v16h a = *(const v16h*)&h16[l15 * H_ + k0 + lhalf * 16];
      for (int g = 0; g < 4; ++g)
        for (int jt = 0; jt < 2; ++jt) {
          const int nn = g * H_ + jcol0 + jt * 16 + l15;
          const v16h b = *(const v16h*)&recT[nn * LDR + lhalf * 16];
          acc[g][jt] = __builtin_amdgcn_wmma_f32_16x16x32_f16(
              false, a, false, b, (short)0, acc[g][jt], false, false);
        }
      __syncthreads();
    }

    // gate combine — all in registers (Keras order i,f,c,o; relu cell act)
    for (int jt = 0; jt < 2; ++jt) {
      const int cb = jt * 16 + l15;          // col within wave's 32-block
      const int col = jcol0 + cb;            // true h column
      const int pcol = jcol0 + permk(cb);    // permuted position for h16
      for (int r = 0; r < 8; ++r) {
        const int m = r + lhalf * 8;
        float zi = acc[0][jt][r], zf = acc[1][jt][r];
        float zc = acc[2][jt][r], zo = acc[3][jt][r];
        float ig = 1.f / (1.f + __expf(-zi));
        float fg = 1.f / (1.f + __expf(-zf));
        float og = 1.f / (1.f + __expf(-zo));
        float cc = fg * creg[jt * 8 + r] + ig * fmaxf(zc, 0.f);
        creg[jt * 8 + r] = cc;
        float hh = og * fmaxf(cc, 0.f);
        h16[m * H_ + pcol] = (_Float16)hh;                       // permuted
        hout[((size_t)(b0 + m) * S_ + t) * H_ + col] = hh;       // natural
      }
    }
    __syncthreads();
  }
}

// ---------------------------------------------------------------------------
// logits[b, 0:2] = sum_i (h_f+h_b)[b,i] * Wd[i, :] + bd
// ---------------------------------------------------------------------------
__global__ __launch_bounds__(256) void k_final(
    const float* __restrict__ h_f, const float* __restrict__ h_b,
    const float* __restrict__ Wd, const float* __restrict__ bd,
    float* __restrict__ out) {
  __shared__ float red0[256];
  __shared__ float red1[256];
  const int b = blockIdx.x;
  const int tid = threadIdx.x;
  float p0 = 0.f, p1 = 0.f;
  const size_t base = (size_t)b * (S_ * H_);
  for (int i = tid; i < S_ * H_; i += 256) {
    float v = h_f[base + i] + h_b[base + i];
    p0 += v * Wd[(size_t)i * 2 + 0];
    p1 += v * Wd[(size_t)i * 2 + 1];
  }
  red0[tid] = p0; red1[tid] = p1;
  __syncthreads();
  for (int s = 128; s > 0; s >>= 1) {
    if (tid < s) { red0[tid] += red0[tid + s]; red1[tid] += red1[tid + s]; }
    __syncthreads();
  }
  if (tid == 0) {
    out[b * 2 + 0] = red0[0] + bd[0];
    out[b * 2 + 1] = red1[0] + bd[1];
  }
}

// ---------------------------------------------------------------------------
extern "C" void kernel_launch(void* const* d_in, const int* in_sizes, int n_in,
                              void* d_out, int out_size, void* d_ws,
                              size_t ws_size, hipStream_t stream) {
  (void)in_sizes; (void)n_in; (void)out_size; (void)ws_size;
  const float* x        = (const float*)d_in[0];
  const float* kernel_f = (const float*)d_in[1];
  const float* rec_f    = (const float*)d_in[2];
  const float* bias_f   = (const float*)d_in[3];
  const float* kernel_b = (const float*)d_in[4];
  const float* rec_b    = (const float*)d_in[5];
  const float* bias_b   = (const float*)d_in[6];
  const float* Wd       = (const float*)d_in[7];
  const float* bd       = (const float*)d_in[8];
  float* out = (float*)d_out;

  char* ws = (char*)d_ws;
  size_t off = 0;
  _Float16* x16 = (_Float16*)(ws + off); off += (size_t)B_ * S_ * E_ * 2;
  _Float16* kT  = (_Float16*)(ws + off); off += (size_t)2 * G4 * E_ * 2;
  _Float16* rT  = (_Float16*)(ws + off); off += (size_t)2 * G4 * H_ * 2;
  _Float16* xz_f = (_Float16*)(ws + off); off += (size_t)B_ * S_ * G4 * 2;
  _Float16* xz_b = (_Float16*)(ws + off); off += (size_t)B_ * S_ * G4 * 2;
  float* h_f = (float*)(ws + off); off += (size_t)B_ * S_ * H_ * 4;
  float* h_b = (float*)(ws + off); off += (size_t)B_ * S_ * H_ * 4;

  // 1. convert x; transpose+permute weights into fragment-order f16
  k_f32_to_f16<<<2048, 256, 0, stream>>>(x, x16, B_ * S_ * E_);
  k_w_to_f16T<<<1024, 256, 0, stream>>>(kernel_f, kT, E_);
  k_w_to_f16T<<<1024, 256, 0, stream>>>(kernel_b, kT + (size_t)G4 * E_, E_);
  k_w_to_f16T<<<512, 256, 0, stream>>>(rec_f, rT, H_);
  k_w_to_f16T<<<512, 256, 0, stream>>>(rec_b, rT + (size_t)G4 * H_, H_);

  // 2. input projection GEMM (both directions)
  dim3 g2(G4 / 64, (B_ * S_) / 64, 2);
  k_gemm_xz<<<g2, 128, 0, stream>>>(x16, kT, bias_f, bias_b, xz_f, xz_b);

  // 3. recurrent scan: 4 batch tiles x 2 directions, persistent
  size_t lstm_lds = 16 * 256 * 2 + 1024 * 48 * 2;  // 8 KB + 96 KB = 106496 B
  dim3 g3(4, 2);
  k_lstm<<<g3, 256, lstm_lds, stream>>>(rT, xz_f, xz_b, h_f, h_b);

  // 4. dense head
  k_final<<<B_, 256, 0, stream>>>(h_f, h_b, Wd, bd, out);
}
